// WaveNet_86208583565441
// MI455X (gfx1250) — matrix operand order repaired and checked by reading, not compiled, not run on hardware
//
#include <hip/hip_runtime.h>

// ---------------------------------------------------------------------------
// WaveNet forward for MI455X (gfx1250, wave32, WMMA bf16 16x16x32, f32 accum)
//
// Activations: [M = B*T, C] row-major bf16 (time shift == row shift).
// Weights: converted once per launch to bf16, [C_out][C_in] row-major, which
// is the B-operand layout load_b uses (lane = N, 16 contiguous K).
// LDS staging of activation tiles uses GLOBAL_LOAD_ASYNC_TO_LDS_B128
// (ASYNCcnt-tracked, no VGPR round-trip); causally-masked halo rows are
// zero-filled with plain ds stores under divergent EXEC.
// ---------------------------------------------------------------------------

typedef __attribute__((ext_vector_type(16))) __bf16 bf16x16;
typedef __attribute__((ext_vector_type(8)))  __bf16 bf16x8;
typedef __attribute__((ext_vector_type(8)))  float  f32x8;
typedef unsigned short u16;
typedef unsigned int   u32;
typedef unsigned long long u64;

#define TLEN 8192
#define NB   16
#define MTOT (TLEN * NB)   // 131072 rows
#define CE   256
#define CR   128
#define GC   64
#define CS   256
#define NV   256
#define NL   12
#define M_TILE 128
#define NBLK (MTOT / M_TILE)   // 1024 workgroups

#define WMMA_BF16(a, b, c) \
  __builtin_amdgcn_wmma_f32_16x16x32_bf16(false, (a), false, (b), (short)0, (c), false, false)

__device__ __forceinline__ u16 f2bf(float f) {
  u32 u = __float_as_uint(f);
  u32 r = (u + 0x7FFFu + ((u >> 16) & 1u)) >> 16;   // round-to-nearest-even
  return (u16)r;
}
__device__ __forceinline__ float bf2f(u16 h) {
  return __uint_as_float(((u32)h) << 16);
}
__device__ __forceinline__ f32x8 zero8() {
  f32x8 z;
#pragma unroll
  for (int i = 0; i < 8; ++i) z[i] = 0.0f;
  return z;
}

// Async copy 16B global -> LDS (per-lane, EXEC-predicated by caller's control flow)
__device__ __forceinline__ void async_b128(void* lds_dst, const void* gsrc) {
  u32 ldsa = (u32)(size_t)lds_dst;   // generic LDS addr low 32 bits == LDS offset
  asm volatile("global_load_async_to_lds_b128 %0, %1, off"
               :: "v"(ldsa), "v"(gsrc) : "memory");
}
__device__ __forceinline__ void wait_async() {
  asm volatile("s_wait_asynccnt 0x0" ::: "memory");
}

// A fragment (16x32 bf16): row-major source (LDS), row stride lda (elements).
// lane m = lane&15 ; k-base = 16*(lane>=16); elems 0-15 -> k0+base+e (contiguous)
__device__ __forceinline__ bf16x16 load_a(const u16* src, int lda, int row0, int k0) {
  int lane = threadIdx.x & 31;
  int m    = lane & 15;
  int kb   = (lane >> 4) * 16;
  const __bf16* p = (const __bf16*)src + (row0 + m) * lda + k0 + kb;
  bf16x8 lo = *(const bf16x8*)(p);
  bf16x8 hi = *(const bf16x8*)(p + 8);
  return __builtin_shufflevector(lo, hi, 0,1,2,3,4,5,6,7,8,9,10,11,12,13,14,15);
}

// B fragment (32x16 bf16): weights [C_out][C_in] row-major in global.
// lane n = lane&15 ; k = 16*(lane>=16) + e, 16 contiguous bf16 (one 32B load)
__device__ __forceinline__ bf16x16 load_b(const u16* w, int ldw, int n0, int k0) {
  int lane = threadIdx.x & 31;
  int n    = lane & 15;
  int kb   = (lane >> 4) * 16;
  return *(const bf16x16*)((const __bf16*)w + (n0 + n) * ldw + k0 + kb);
}

// ---------------------------------------------------------------------------
// Weight conversion kernels
// ---------------------------------------------------------------------------
__global__ void cvt_f32_bf16(const float* __restrict__ src, u16* __restrict__ dst, int n) {
  int i = blockIdx.x * 256 + threadIdx.x;
  if (i < n) dst[i] = f2bf(src[i]);
}

// src [L][O][I][2] (conv weights, K innermost) -> dst [L][2][O][I] (tap-major)
__global__ void cvt_conv_w(const float* __restrict__ src, u16* __restrict__ dst,
                           int Lc, int O, int I) {
  int i = blockIdx.x * 256 + threadIdx.x;
  int total = Lc * O * I * 2;
  if (i < total) {
    int k  = i & 1;
    int ii = (i >> 1) % I;
    int o  = ((i >> 1) / I) % O;
    int l  = i / (2 * I * O);
    dst[(((size_t)l * 2 + k) * O + o) * I + ii] = f2bf(src[i]);
  }
}

// ---------------------------------------------------------------------------
// Fused embedding gather + initial causal conv (K=2, dilation 1)
//   x[n] = W0 @ emb[tok[n-1]] + W1 @ emb[tok[n]] + b      -> bf16 [M][CR]
// ---------------------------------------------------------------------------
__global__ __launch_bounds__(256) void wavenet_init(
    const int* __restrict__ tokens, const u16* __restrict__ ebf,
    const u16* __restrict__ winit /*[2][CR][CE]*/, const float* __restrict__ binit,
    u16* __restrict__ xout) {
  extern __shared__ char smem[];
  const int LDAH = CE + 8;                 // 264, row = 528B (16B aligned)
  u16* ht = (u16*)smem;                    // [129][264] rows m0-1 .. m0+127

  const int tid  = threadIdx.x;
  const int wave = tid >> 5;
  const int m0   = blockIdx.x * M_TILE;
  const int bst  = (m0 / TLEN) * TLEN;

  // prefetch the init conv weights (2*128*256*2 = 128KB) while gathers run
  for (int off = tid * 128; off < 2 * CR * CE * 2; off += 256 * 128)
    __builtin_prefetch((const char*)winit + off, 0, 1);

  // async-gather embeddings (129 rows x 256ch, 16B chunks)
  for (int i = tid; i < 129 * (CE / 8); i += 256) {
    int r = i >> 5;
    int c = (i & 31) * 8;
    long gi = (long)m0 - 1 + r;
    u16* lp = ht + r * LDAH + c;
    if (gi >= bst) {
      int tok = tokens[gi];
      async_b128(lp, ebf + (size_t)tok * CE + c);
    } else {
      *(uint4*)lp = make_uint4(0u, 0u, 0u, 0u);
    }
  }
  wait_async();
  __syncthreads();

  f32x8 acc[8];
#pragma unroll
  for (int mt = 0; mt < 8; ++mt) acc[mt] = zero8();

  const int ncol = wave;                   // output-channel block (N = CR = 128)
#pragma unroll
  for (int tap = 0; tap < 2; ++tap) {      // tap0 -> x[t-1] (lds row m), tap1 -> x[t] (row m+1)
    const u16* wt = winit + (size_t)tap * CR * CE;
#pragma unroll
    for (int ks = 0; ks < CE; ks += 32) {
      bf16x16 b = load_b(wt, CE, ncol * 16, ks);
#pragma unroll
      for (int mt = 0; mt < 8; ++mt) {
        bf16x16 a = load_a(ht, LDAH, mt * 16 + tap, ks);
        acc[mt] = WMMA_BF16(a, b, acc[mt]);
      }
    }
  }

  int lane = tid & 31, n = lane & 15, mofs = (lane >> 4) * 8;
  float bias = binit[ncol * 16 + n];
#pragma unroll
  for (int mt = 0; mt < 8; ++mt)
#pragma unroll
    for (int r = 0; r < 8; ++r) {
      long row = m0 + mt * 16 + mofs + r;
      xout[row * CR + ncol * 16 + n] = f2bf(acc[mt][r] + bias);
    }
}

// ---------------------------------------------------------------------------
// Fused residual layer: dilated conv (K=2, dil 2) -> gate -> skip GEMM (+=)
// -> res GEMM (+x).  One 128-row tile per workgroup, 8 waves.
// ---------------------------------------------------------------------------
__global__ __launch_bounds__(256) void wavenet_layer(
    const u16* __restrict__ xin, u16* __restrict__ xout, float* __restrict__ skip,
    const u16* __restrict__ wdil /*[2][CR][CR]*/, const float* __restrict__ bdil,
    const u16* __restrict__ wskip /*[CS][GC]*/,   const float* __restrict__ bskip,
    const u16* __restrict__ wres  /*[CR][GC]*/,   const float* __restrict__ bres,
    int first) {
  extern __shared__ char smem[];
  const int LDAX = CR + 8;   // 136 (row = 272B, 16B aligned)
  const int LDXC = CR + 4;   // 132
  const int LDG  = GC + 8;   // 72
  u16*   xt = (u16*)smem;                                     // [130][136] bf16
  float* xc = (float*)(smem + 130 * LDAX * 2);                // [128][132] f32
  u16*   g  = (u16*)(smem + 130 * LDAX * 2 + 128 * LDXC * 4); // [128][72] bf16

  const int tid  = threadIdx.x;
  const int wave = tid >> 5;
  const int m0   = blockIdx.x * M_TILE;
  const int bst  = (m0 / TLEN) * TLEN;
  const int lane = tid & 31;
  const int n    = lane & 15;
  const int mofs = (lane >> 4) * 8;

  // prefetch this layer's weights (dil 64KB + skip 32KB + res 16KB)
  for (int off = tid * 128; off < 2 * CR * CR * 2; off += 256 * 128)
    __builtin_prefetch((const char*)wdil + off, 0, 1);
  for (int off = tid * 128; off < CS * GC * 2; off += 256 * 128)
    __builtin_prefetch((const char*)wskip + off, 0, 1);
  for (int off = tid * 128; off < CR * GC * 2; off += 256 * 128)
    __builtin_prefetch((const char*)wres + off, 0, 1);

  // async-stage x rows m0-2 .. m0+127 (130 rows), zero-masked at batch start
  for (int i = tid; i < 130 * (CR / 8); i += 256) {
    int r = i >> 4;
    int c = (i & 15) * 8;
    long gi = (long)m0 - 2 + r;
    u16* lp = xt + r * LDAX + c;
    if (gi >= bst) {
      async_b128(lp, xin + gi * CR + c);
    } else {
      *(uint4*)lp = make_uint4(0u, 0u, 0u, 0u);
    }
  }
  wait_async();
  __syncthreads();

  // ---- dilated conv GEMM: xc = W0 @ x[t-2] + W1 @ x[t] ----
  f32x8 acc[8];
#pragma unroll
  for (int mt = 0; mt < 8; ++mt) acc[mt] = zero8();
  const int ncol = wave;
#pragma unroll
  for (int tap = 0; tap < 2; ++tap) {      // tap0: lds row m (=x[t-2]); tap1: row m+2 (=x[t])
    const u16* wt = wdil + (size_t)tap * CR * CR;
#pragma unroll
    for (int ks = 0; ks < CR; ks += 32) {
      bf16x16 b = load_b(wt, CR, ncol * 16, ks);
#pragma unroll
      for (int mt = 0; mt < 8; ++mt) {
        bf16x16 a = load_a(xt, LDAX, mt * 16 + tap * 2, ks);
        acc[mt] = WMMA_BF16(a, b, acc[mt]);
      }
    }
  }
  {
    float bias = bdil[ncol * 16 + n];
#pragma unroll
    for (int mt = 0; mt < 8; ++mt)
#pragma unroll
      for (int r = 0; r < 8; ++r)
        xc[(mt * 16 + mofs + r) * LDXC + ncol * 16 + n] = acc[mt][r] + bias;
  }
  __syncthreads();

  // ---- gate: g = tanh(xc[:64]) * sigmoid(xc[64:]) -> bf16 ----
  for (int i = tid; i < 128 * GC; i += 256) {
    int r = i >> 6, c = i & 63;
    float ta = tanhf(xc[r * LDXC + c]);
    float sg = 1.0f / (1.0f + __expf(-xc[r * LDXC + GC + c]));
    g[r * LDG + c] = f2bf(ta * sg);
  }
  __syncthreads();

  // ---- skip GEMM: skip[m0..][0..255] += g @ w_skip^T + b_skip ----
#pragma unroll
  for (int half = 0; half < 2; ++half) {
    int nt = wave + half * 8;
    f32x8 sa[8];
#pragma unroll
    for (int mt = 0; mt < 8; ++mt) sa[mt] = zero8();
#pragma unroll
    for (int ks = 0; ks < GC; ks += 32) {
      bf16x16 b = load_b(wskip, GC, nt * 16, ks);
#pragma unroll
      for (int mt = 0; mt < 8; ++mt) {
        bf16x16 a = load_a(g, LDG, mt * 16, ks);
        sa[mt] = WMMA_BF16(a, b, sa[mt]);
      }
    }
    float bias = bskip[nt * 16 + n];
#pragma unroll
    for (int mt = 0; mt < 8; ++mt)
#pragma unroll
      for (int r = 0; r < 8; ++r) {
        long row = m0 + mt * 16 + mofs + r;
        float* p = skip + row * CS + nt * 16 + n;
        float v = sa[mt][r] + bias;
        if (first) *p = v; else *p += v;
      }
  }

  // ---- res GEMM: x_out = g @ w_res^T + b_res + x ----
  {
    f32x8 ra[8];
#pragma unroll
    for (int mt = 0; mt < 8; ++mt) ra[mt] = zero8();
#pragma unroll
    for (int ks = 0; ks < GC; ks += 32) {
      bf16x16 b = load_b(wres, GC, wave * 16, ks);
#pragma unroll
      for (int mt = 0; mt < 8; ++mt) {
        bf16x16 a = load_a(g, LDG, mt * 16, ks);
        ra[mt] = WMMA_BF16(a, b, ra[mt]);
      }
    }
    float bias = bres[wave * 16 + n];
#pragma unroll
    for (int mt = 0; mt < 8; ++mt)
#pragma unroll
      for (int r = 0; r < 8; ++r) {
        int mr = mt * 16 + mofs + r;
        float xv = bf2f(xt[(mr + 2) * LDAX + wave * 16 + n]);   // unshifted x
        xout[((long)m0 + mr) * CR + wave * 16 + n] = f2bf(ra[mt][r] + bias + xv);
      }
  }
}

// ---------------------------------------------------------------------------
// Output head: s = relu(skip/12); s = relu(s @ w1^T + b1); out = s @ w2^T + b2
// Writes d_out in [B, V, T] layout (per-lane float4 pairs, contiguous in T).
// ---------------------------------------------------------------------------
__global__ __launch_bounds__(256) void wavenet_out(
    const float* __restrict__ skip,
    const u16* __restrict__ w1, const float* __restrict__ b1,
    const u16* __restrict__ w2, const float* __restrict__ b2,
    float* __restrict__ out) {
  extern __shared__ char smem[];
  const int LDS1 = CS + 8;   // 264
  u16* sb = (u16*)smem;                       // [128][264]
  u16* s2 = (u16*)(smem + 128 * LDS1 * 2);    // [128][264]

  const int tid  = threadIdx.x;
  const int wave = tid >> 5;
  const int m0   = blockIdx.x * M_TILE;
  const float inv = 1.0f / (float)NL;
  const int lane = tid & 31;
  const int n    = lane & 15;
  const int mofs = (lane >> 4) * 8;

  for (int off = tid * 128; off < CS * CS * 2; off += 256 * 128)
    __builtin_prefetch((const char*)w1 + off, 0, 1);
  for (int off = tid * 128; off < NV * CS * 2; off += 256 * 128)
    __builtin_prefetch((const char*)w2 + off, 0, 1);

  // stage relu(skip/12) as bf16
  for (int i = tid; i < 128 * (CS / 4); i += 256) {
    int r = i >> 6, c4 = (i & 63) * 4;
    float4 v = *(const float4*)(skip + ((long)m0 + r) * CS + c4);
    u64 pack = (u64)f2bf(fmaxf(v.x * inv, 0.0f)) |
               ((u64)f2bf(fmaxf(v.y * inv, 0.0f)) << 16) |
               ((u64)f2bf(fmaxf(v.z * inv, 0.0f)) << 32) |
               ((u64)f2bf(fmaxf(v.w * inv, 0.0f)) << 48);
    *(u64*)(sb + r * LDS1 + c4) = pack;
  }
  __syncthreads();

  // GEMM1: [128x256] @ [256x256] -> relu -> s2 (bf16)
#pragma unroll
  for (int half = 0; half < 2; ++half) {
    int nt = wave + half * 8;
    f32x8 a1[8];
#pragma unroll
    for (int mt = 0; mt < 8; ++mt) a1[mt] = zero8();
#pragma unroll
    for (int ks = 0; ks < CS; ks += 32) {
      bf16x16 b = load_b(w1, CS, nt * 16, ks);
#pragma unroll
      for (int mt = 0; mt < 8; ++mt) {
        bf16x16 a = load_a(sb, LDS1, mt * 16, ks);
        a1[mt] = WMMA_BF16(a, b, a1[mt]);
      }
    }
    float bias = b1[nt * 16 + n];
#pragma unroll
    for (int mt = 0; mt < 8; ++mt)
#pragma unroll
      for (int r = 0; r < 8; ++r)
        s2[(mt * 16 + mofs + r) * LDS1 + nt * 16 + n] = f2bf(fmaxf(a1[mt][r] + bias, 0.0f));
  }
  __syncthreads();

  // GEMM2: [128x256] @ [256x256] -> d_out[B][V][T]
  const int bidx  = m0 / TLEN;
  const int tbase = m0 % TLEN;
#pragma unroll
  for (int half = 0; half < 2; ++half) {
    int nt = wave + half * 8;
    f32x8 a2[8];
#pragma unroll
    for (int mt = 0; mt < 8; ++mt) a2[mt] = zero8();
#pragma unroll
    for (int ks = 0; ks < CS; ks += 32) {
      bf16x16 b = load_b(w2, CS, nt * 16, ks);
#pragma unroll
      for (int mt = 0; mt < 8; ++mt) {
        bf16x16 a = load_a(s2, LDS1, mt * 16, ks);
        a2[mt] = WMMA_BF16(a, b, a2[mt]);
      }
    }
    int vch = nt * 16 + n;
    float bias = b2[vch];
    float* obase = out + ((long)bidx * NV + vch) * TLEN;
#pragma unroll
    for (int mt = 0; mt < 8; ++mt) {
      int t0 = tbase + mt * 16 + mofs;     // 8 consecutive t per lane
      float4 lo = make_float4(a2[mt][0] + bias, a2[mt][1] + bias,
                              a2[mt][2] + bias, a2[mt][3] + bias);
      float4 hi = make_float4(a2[mt][4] + bias, a2[mt][5] + bias,
                              a2[mt][6] + bias, a2[mt][7] + bias);
      *(float4*)(obase + t0)     = lo;
      *(float4*)(obase + t0 + 4) = hi;
    }
  }
}

// ---------------------------------------------------------------------------
// Host launcher
// ---------------------------------------------------------------------------
extern "C" void kernel_launch(void* const* d_in, const int* in_sizes, int n_in,
                              void* d_out, int out_size, void* d_ws, size_t ws_size,
                              hipStream_t stream) {
  const int*   tokens = (const int*)  d_in[0];
  const float* emb    = (const float*)d_in[1];
  const float* w_init = (const float*)d_in[2];
  const float* b_init = (const float*)d_in[3];
  const float* w_dil  = (const float*)d_in[4];
  const float* b_dil  = (const float*)d_in[5];
  const float* w_res  = (const float*)d_in[6];
  const float* b_res  = (const float*)d_in[7];
  const float* w_skip = (const float*)d_in[8];
  const float* b_skip = (const float*)d_in[9];
  const float* w_out1 = (const float*)d_in[10];
  const float* b_out1 = (const float*)d_in[11];
  const float* w_out2 = (const float*)d_in[12];
  const float* b_out2 = (const float*)d_in[13];
  float* outp = (float*)d_out;

  char* ws = (char*)d_ws;
  size_t off = 0;
  auto carve = [&](size_t bytes) -> char* {
    char* p = ws + off;
    off += (bytes + 255) & ~(size_t)255;
    return p;
  };
  u16*   ebf    = (u16*)  carve((size_t)256 * 256 * 2);
  u16*   wbinit = (u16*)  carve((size_t)2 * CR * CE * 2);
  u16*   wbdil  = (u16*)  carve((size_t)NL * 2 * CR * CR * 2);
  u16*   wbskip = (u16*)  carve((size_t)NL * CS * GC * 2);
  u16*   wbres  = (u16*)  carve((size_t)NL * CR * GC * 2);
  u16*   wb1    = (u16*)  carve((size_t)CS * CS * 2);
  u16*   wb2    = (u16*)  carve((size_t)NV * CS * 2);
  u16*   x0     = (u16*)  carve((size_t)MTOT * CR * 2);
  u16*   x1     = (u16*)  carve((size_t)MTOT * CR * 2);
  float* skip   = (float*)carve((size_t)MTOT * CS * 4);
  (void)ws_size; (void)n_in; (void)in_sizes; (void)out_size;

  auto grid1d = [](int n) { return dim3((unsigned)((n + 255) / 256)); };

  // weight conversion (f32 -> bf16), conv weights reordered tap-major
  cvt_f32_bf16<<<grid1d(256 * 256), 256, 0, stream>>>(emb, ebf, 256 * 256);
  cvt_conv_w  <<<grid1d(CR * CE * 2), 256, 0, stream>>>(w_init, wbinit, 1, CR, CE);
  cvt_conv_w  <<<grid1d(NL * CR * CR * 2), 256, 0, stream>>>(w_dil, wbdil, NL, CR, CR);
  cvt_f32_bf16<<<grid1d(NL * CS * GC), 256, 0, stream>>>(w_skip, wbskip, NL * CS * GC);
  cvt_f32_bf16<<<grid1d(NL * CR * GC), 256, 0, stream>>>(w_res, wbres, NL * CR * GC);
  cvt_f32_bf16<<<grid1d(CS * CS), 256, 0, stream>>>(w_out1, wb1, CS * CS);
  cvt_f32_bf16<<<grid1d(NV * CS), 256, 0, stream>>>(w_out2, wb2, NV * CS);

  // embedding + init conv
  {
    size_t shm = (size_t)129 * (CE + 8) * 2;   // ~68 KB
    wavenet_init<<<dim3(NBLK), 256, shm, stream>>>(tokens, ebf, wbinit, b_init, x0);
  }

  // 12 residual layers (ping-pong x buffers; layer 0 stores skip, others add)
  {
    size_t shm = (size_t)130 * (CR + 8) * 2 + (size_t)128 * (CR + 4) * 4 +
                 (size_t)128 * (GC + 8) * 2;   // ~121 KB
    for (int l = 0; l < NL; ++l) {
      const u16* xi = (l & 1) ? x1 : x0;
      u16*       xo = (l & 1) ? x0 : x1;
      wavenet_layer<<<dim3(NBLK), 256, shm, stream>>>(
          xi, xo, skip,
          wbdil + (size_t)l * 2 * CR * CR, b_dil + (size_t)l * CR,
          wbskip + (size_t)l * CS * GC,    b_skip + (size_t)l * CS,
          wbres + (size_t)l * CR * GC,     b_res + (size_t)l * CR,
          (l == 0) ? 1 : 0);
    }
  }

  // output head
  {
    size_t shm = (size_t)2 * 128 * (CS + 8) * 2;   // ~132 KB
    wavenet_out<<<dim3(NBLK), 256, shm, stream>>>(skip, wb1, b_out1, wb2, b_out2, outp);
  }
}
